// SAModule_27298812133727
// MI455X (gfx1250) — matrix operand order repaired and checked by reading, not compiled, not run on hardware
//
#include <hip/hip_runtime.h>

typedef __attribute__((ext_vector_type(16))) _Float16 v16h;
typedef __attribute__((ext_vector_type(8)))  _Float16 v8h;
typedef __attribute__((ext_vector_type(8)))  float    v8f;
typedef __attribute__((ext_vector_type(4)))  float    v4f;

#define BDIM 8
#define CDIM 256
#define NDIM 4096
#define CQK  32
#define MROW 320   // 32 q rows + 32 k rows + 256 v rows

#define WMMA_F16(a, b, c) \
  __builtin_amdgcn_wmma_f32_16x16x32_f16(false, (a), false, (b), (short)0, (c), false, false)
#define CAT16(lo, hi) \
  __builtin_shufflevector((lo), (hi), 0,1,2,3,4,5,6,7,8,9,10,11,12,13,14,15)

// ---------------- workspace layout (bytes) ----------------
constexpr size_t XT_OFF   = 0;                                          // [B][N][C]  f16
constexpr size_t W_OFF    = XT_OFF   + (size_t)BDIM*NDIM*CDIM*2;        // [320][256] f16
constexpr size_t BIAS_OFF = W_OFF    + (size_t)MROW*CDIM*2;             // [320]      f32
constexpr size_t QH_OFF   = BIAS_OFF + (size_t)MROW*4;                  // [B][N][32] f16
constexpr size_t KH_OFF   = QH_OFF   + (size_t)BDIM*NDIM*CQK*2;         // [B][N][32] f16
constexpr size_t VH_OFF   = KH_OFF   + (size_t)BDIM*NDIM*CQK*2;         // [B][C][N]  f16 (channel-major!)
constexpr size_t RM_OFF   = VH_OFF   + (size_t)BDIM*NDIM*CDIM*2;        // [B][N]     f32
constexpr size_t RI_OFF   = RM_OFF   + (size_t)BDIM*NDIM*4;             // [B][N]     f32
// total ~38.2 MB

// in-16-lane-group reductions (wave32; xor of bits 0..3 stays in the half)
__device__ __forceinline__ float rowmax16(float v) {
  #pragma unroll
  for (int m = 1; m < 16; m <<= 1) v = fmaxf(v, __shfl_xor(v, m, 32));
  return v;
}
__device__ __forceinline__ float rowsum16(float v) {
  #pragma unroll
  for (int m = 1; m < 16; m <<= 1) v += __shfl_xor(v, m, 32);
  return v;
}

// CDNA5 async global->LDS DMA (ASYNCcnt-tracked), 16B per lane.
// fills LDS address lds_off from global address gaddr (GV mode, per ISA 15.18.3 op 98)
__device__ __forceinline__ void async_copy_b128(uint32_t lds_off, const void* gaddr) {
  asm volatile("global_load_async_to_lds_b128 %0, %1, off"
               :: "v"(lds_off), "v"((unsigned long long)(uintptr_t)gaddr)
               : "memory");
}
__device__ __forceinline__ void wait_asynccnt0() {
  asm volatile("s_wait_asynccnt 0x0" ::: "memory");
}

// ---------------- kernel 0a: x [B][C][N] f32 -> xt [B][N][C] f16 ----------------
__global__ void __launch_bounds__(256) prep_x_kernel(const float* __restrict__ feat,
                                                     _Float16* __restrict__ xt) {
  int id = blockIdx.x * 256 + threadIdx.x;          // 1,048,576 chunks of 8 pixels
  int nc = id & 511;
  int c  = (id >> 9) & 255;
  int b  = id >> 17;
  const float* src = feat + ((size_t)b * CDIM + c) * NDIM + nc * 8;
  float4 f0 = *(const float4*)src;
  float4 f1 = *(const float4*)(src + 4);
  _Float16* dst = xt + ((size_t)b * NDIM + (size_t)nc * 8) * CDIM + c;
  dst[0*CDIM] = (_Float16)f0.x; dst[1*CDIM] = (_Float16)f0.y;
  dst[2*CDIM] = (_Float16)f0.z; dst[3*CDIM] = (_Float16)f0.w;
  dst[4*CDIM] = (_Float16)f1.x; dst[5*CDIM] = (_Float16)f1.y;
  dst[6*CDIM] = (_Float16)f1.z; dst[7*CDIM] = (_Float16)f1.w;
}

// ---------------- kernel 0b: pack weights/biases ----------------
__global__ void __launch_bounds__(256) prep_w_kernel(
    const float* __restrict__ w1, const float* __restrict__ w2, const float* __restrict__ w3,
    const float* __restrict__ b1, const float* __restrict__ b2, const float* __restrict__ b3,
    _Float16* __restrict__ W_all, float* __restrict__ bias_all) {
  int row = blockIdx.x, col = threadIdx.x;
  float v;
  if (row < 32)       v = w1[row * CDIM + col];
  else if (row < 64)  v = w2[(row - 32) * CDIM + col];
  else                v = w3[(row - 64) * CDIM + col];
  W_all[(size_t)row * CDIM + col] = (_Float16)v;
  if (blockIdx.x == 0) {
    for (int i = threadIdx.x; i < MROW; i += 256)
      bias_all[i] = (i < 32) ? b1[i] : (i < 64) ? b2[i - 32] : b3[i - 64];
  }
}

// ---------------- kernel 1: QKV projection  Y[320,4096] = W_all @ xt(b)^T ----------------
// block = 256 thr (8 waves), handles 64 pixels x 320 outputs; 10 16x16 tiles per wave.
// q,k stored [B][N][32] (pixel-major, packed 16B chunks); v stored [B][C][N] (channel-major)
__global__ void __launch_bounds__(256) qkv_proj_kernel(
    const _Float16* __restrict__ W_all, const float* __restrict__ bias_all,
    const _Float16* __restrict__ xt,
    _Float16* __restrict__ qh, _Float16* __restrict__ kh, _Float16* __restrict__ vh) {
  int tid = threadIdx.x, lane = tid & 31, wave = tid >> 5;
  int b  = blockIdx.x >> 6;
  int n0 = (blockIdx.x & 63) << 6;
  int nk = lane & 15, hi = lane >> 4;
  const _Float16* xb = xt + (size_t)b * NDIM * CDIM;

  for (int i = 0; i < 10; ++i) {
    int t  = wave * 10 + i;
    int mt = t % 20;           // output row tile (0..19)
    int nt = t / 20;           // pixel tile (0..3)
    int pix = n0 + nt * 16 + nk;
    v8f acc = {};
    #pragma unroll
    for (int ks = 0; ks < CDIM; ks += 32) {
      const _Float16* wrow = W_all + (size_t)(mt * 16 + nk) * CDIM + ks;
      v8h a0 = *(const v8h*)(wrow + hi * 8);
      v8h a1 = *(const v8h*)(wrow + 16 + hi * 8);
      v16h af = CAT16(a0, a1);
      v16h bf = CAT16(*(const v8h*)(xb + (size_t)pix * CDIM + ks + hi * 8), *(const v8h*)(xb + (size_t)pix * CDIM + ks + 16 + hi * 8));
      acc = WMMA_F16(af, bf, acc);
    }
    v8h hv;
    #pragma unroll
    for (int r = 0; r < 8; ++r)
      hv[r] = (_Float16)(acc[r] + bias_all[mt * 16 + 8 * hi + r]);
    if (mt < 4) {  // q (mt 0,1) / k (mt 2,3), layout [B][N][32]
      _Float16* dst = (mt < 2) ? qh : kh;
      int d0 = (mt & 1) * 16 + 8 * hi;
      *(v8h*)(dst + ((size_t)b * NDIM + pix) * CQK + d0) = hv;
    } else {       // v, layout [B][C][N]: 8 consecutive channels, fixed pixel
      int c0 = (mt - 4) * 16 + 8 * hi;
      _Float16* vdst = vh + (size_t)b * CDIM * NDIM + pix;
      #pragma unroll
      for (int r = 0; r < 8; ++r)
        vdst[(size_t)(c0 + r) * NDIM] = hv[r];
    }
  }
}

// ---------------- kernel 2: softmax row stats (online max/sum over all 4096 keys) --------
// one wave per 16-query tile; 1 WMMA per 16-key block (K = head dim = 32).
__global__ void __launch_bounds__(256) attn_stats_kernel(
    const _Float16* __restrict__ qh, const _Float16* __restrict__ kh,
    float* __restrict__ rowmax, float* __restrict__ rowinv) {
  int tid = threadIdx.x, lane = tid & 31, wave = tid >> 5;
  int tile = blockIdx.x * 8 + wave;     // 0..2047
  int b  = tile >> 8;
  int q0 = (tile & 255) << 4;
  int nk = lane & 15, hi = lane >> 4;

  const _Float16* qbase = qh + ((size_t)b * NDIM + q0) * CQK;
  v8h qa = *(const v8h*)(qbase + (size_t)nk * CQK + hi * 8);
  v8h qb = *(const v8h*)(qbase + (size_t)nk * CQK + 16 + hi * 8);
  v16h qf = CAT16(qa, qb);

  float m_run[8], l_run[8];
  #pragma unroll
  for (int r = 0; r < 8; ++r) { m_run[r] = -1e30f; l_run[r] = 0.0f; }

  const _Float16* kbase = kh + (size_t)b * NDIM * CQK;
  for (int kb = 0; kb < NDIM; kb += 16) {
    v16h kf = CAT16(*(const v8h*)(kbase + (size_t)(kb + nk) * CQK + hi * 8), *(const v8h*)(kbase + (size_t)(kb + nk) * CQK + 16 + hi * 8));
    v8f s = {};
    s = WMMA_F16(qf, kf, s);
    #pragma unroll
    for (int r = 0; r < 8; ++r) {
      float sv = s[r];
      float bm = rowmax16(sv);
      float nm = fmaxf(m_run[r], bm);
      float p  = __expf(sv - nm);
      float rs = rowsum16(p);
      l_run[r] = l_run[r] * __expf(m_run[r] - nm) + rs;
      m_run[r] = nm;
    }
  }
  float mm = m_run[0], ll = l_run[0];
  #pragma unroll
  for (int r = 1; r < 8; ++r) if (nk == r) { mm = m_run[r]; ll = l_run[r]; }
  if (nk < 8) {
    int m = nk + 8 * hi;
    size_t g = (size_t)b * NDIM + q0 + m;
    rowmax[g] = mm;
    rowinv[g] = 1.0f / ll;
  }
}

// ---------------- kernel 3: flash output pass  O = softmax(QK^T) @ V^T ----------------
#define KBLK 64
#define VSTR 80   // f16 stride: rows 160B -> 16/32B aligned chunks, good bank spread
#define PSTR 80

__global__ void __launch_bounds__(256) attn_out_kernel(
    const _Float16* __restrict__ qh, const _Float16* __restrict__ kh,
    const _Float16* __restrict__ vh,
    const float* __restrict__ rowmax, const float* __restrict__ rowinv,
    const float* __restrict__ gamma, float* __restrict__ out) {
  __shared__ _Float16 vbuf[CDIM * VSTR];   // V block [ch][key] (padded), 40 KB
  __shared__ _Float16 pbuf[16 * PSTR];     // P tile [query][key], 2.5 KB

  int tid = threadIdx.x, lane = tid & 31, wave = tid >> 5;
  int b  = blockIdx.x >> 8;
  int q0 = (blockIdx.x & 255) << 4;
  int nk = lane & 15, hi = lane >> 4;

  float rmax[8], rinv[8];
  #pragma unroll
  for (int r = 0; r < 8; ++r) {
    size_t g = (size_t)b * NDIM + q0 + r + 8 * hi;
    rmax[r] = rowmax[g];
    rinv[r] = rowinv[g];
  }

  const _Float16* qbase = qh + ((size_t)b * NDIM + q0) * CQK;
  v8h qa = *(const v8h*)(qbase + (size_t)nk * CQK + hi * 8);
  v8h qb = *(const v8h*)(qbase + (size_t)nk * CQK + 16 + hi * 8);
  v16h qf = CAT16(qa, qb);

  const _Float16* kbase = kh + (size_t)b * NDIM * CQK;
  const _Float16* vbase = vh + (size_t)b * CDIM * NDIM;   // channel-major [C][N]

  uint32_t vbufBase = (uint32_t)(uintptr_t)(&vbuf[0]);

  v8f acc0 = {}, acc1 = {};
  int ct0 = wave * 2;  // each wave owns 2 channel tiles (32 channels)

  for (int kb = 0; kb < NDIM; kb += KBLK) {
    __syncthreads();   // previous iteration consumers done with vbuf/pbuf
    // stage V block [256 ch][64 keys] via async DMA: 2048 chunks of 16B, no transpose
    // needed (channel-major global rows land as padded LDS rows).
    #pragma unroll
    for (int j = 0; j < 8; ++j) {
      int i  = tid * 8 + j;       // consecutive j -> consecutive global 16B chunks
      int c  = i >> 3;            // channel 0..255
      int kc = i & 7;             // key chunk (8 keys = 16B)
      async_copy_b128((uint32_t)(c * VSTR + kc * 8) * 2 + vbufBase,
                      vbase + (size_t)c * NDIM + kb + kc * 8);
    }
    // overlap DMA with S-tile compute:
    // waves 0..3: S tile (16q x 16k) in one WMMA, then P = exp(S - rowmax) -> LDS
    if (wave < 4) {
      v16h kf = CAT16(*(const v8h*)(kbase + (size_t)(kb + wave * 16 + nk) * CQK + hi * 8), *(const v8h*)(kbase + (size_t)(kb + wave * 16 + nk) * CQK + 16 + hi * 8));
      v8f s = {};
      s = WMMA_F16(qf, kf, s);
      #pragma unroll
      for (int r = 0; r < 8; ++r) {
        float p = __expf(s[r] - rmax[r]);
        pbuf[(r + 8 * hi) * PSTR + wave * 16 + nk] = (_Float16)p;
      }
    }
    wait_asynccnt0();  // own DMA landed; barrier makes it workgroup-wide
    __syncthreads();   // pbuf + vbuf ready
    // O += P[16x64] @ V[64 x 32ch(this wave)]
    #pragma unroll
    for (int ks = 0; ks < KBLK; ks += 32) {
      v8h pa = *(const v8h*)(&pbuf[nk * PSTR + ks + hi * 8]);
      v8h pb = *(const v8h*)(&pbuf[nk * PSTR + ks + 16 + hi * 8]);
      v16h pf = CAT16(pa, pb);
      v16h vf0 = CAT16(*(const v8h*)(&vbuf[(ct0 * 16 + nk) * VSTR + ks + hi * 8]), *(const v8h*)(&vbuf[(ct0 * 16 + nk) * VSTR + ks + 16 + hi * 8]));
      acc0 = WMMA_F16(pf, vf0, acc0);
      v16h vf1 = CAT16(*(const v8h*)(&vbuf[((ct0 + 1) * 16 + nk) * VSTR + ks + hi * 8]), *(const v8h*)(&vbuf[((ct0 + 1) * 16 + nk) * VSTR + ks + 16 + hi * 8]));
      acc1 = WMMA_F16(pf, vf1, acc1);
    }
  }

  // finalize: normalize by 1/rowsum, write feat_refine + gamma*feat_refine
  float gm = gamma[0];
  size_t scaledOff = (size_t)BDIM * CDIM * NDIM;
  #pragma unroll
  for (int ti = 0; ti < 2; ++ti) {
    v8f a = ti ? acc1 : acc0;
    int c = (ct0 + ti) * 16 + nk;
    float* dst = out + ((size_t)b * CDIM + c) * NDIM + q0 + 8 * hi;
    v4f f0, f1;
    #pragma unroll
    for (int r = 0; r < 4; ++r) {
      f0[r] = a[r] * rinv[r];
      f1[r] = a[r + 4] * rinv[r + 4];
    }
    *(v4f*)dst = f0;
    *(v4f*)(dst + 4) = f1;
    *(v4f*)(dst + scaledOff) = f0 * gm;
    *(v4f*)(dst + scaledOff + 4) = f1 * gm;
  }
}

// ---------------- host launcher ----------------
extern "C" void kernel_launch(void* const* d_in, const int* in_sizes, int n_in,
                              void* d_out, int out_size, void* d_ws, size_t ws_size,
                              hipStream_t stream) {
  (void)in_sizes; (void)n_in; (void)out_size; (void)ws_size;
  const float* feat  = (const float*)d_in[0];
  const float* w1    = (const float*)d_in[1];
  const float* b1    = (const float*)d_in[2];
  const float* w2    = (const float*)d_in[3];
  const float* b2    = (const float*)d_in[4];
  const float* w3    = (const float*)d_in[5];
  const float* b3    = (const float*)d_in[6];
  const float* gamma = (const float*)d_in[7];

  char* ws = (char*)d_ws;
  _Float16* xt       = (_Float16*)(ws + XT_OFF);
  _Float16* W_all    = (_Float16*)(ws + W_OFF);
  float*    bias_all = (float*)   (ws + BIAS_OFF);
  _Float16* qh       = (_Float16*)(ws + QH_OFF);
  _Float16* kh       = (_Float16*)(ws + KH_OFF);
  _Float16* vh       = (_Float16*)(ws + VH_OFF);
  float*    rmax     = (float*)   (ws + RM_OFF);
  float*    rinv     = (float*)   (ws + RI_OFF);
  float*    out      = (float*)d_out;

  prep_x_kernel<<<4096, 256, 0, stream>>>(feat, xt);
  prep_w_kernel<<<MROW, 256, 0, stream>>>(w1, w2, w3, b1, b2, b3, W_all, bias_all);
  qkv_proj_kernel<<<BDIM * (NDIM / 64), 256, 0, stream>>>(W_all, bias_all, xt, qh, kh, vh);
  attn_stats_kernel<<<BDIM * (NDIM / 16) / 8, 256, 0, stream>>>(qh, kh, rmax, rinv);
  attn_out_kernel<<<BDIM * (NDIM / 16), 256, 0, stream>>>(qh, kh, vh, rmax, rinv, gamma, out);
}
